// Mymodel_89051851915811
// MI455X (gfx1250) — compile-verified, matrix operand order checked by reading, not run on hardware
//
#include <hip/hip_runtime.h>
#include <hip/hip_bf16.h>
#include <stdint.h>

#define NN       20000      // nodes
#define NE       320000     // edges
#define DIN      256
#define DHID     128
#define DEDG     256
#define DOUT     86
#define DOUTP    96         // padded to 6 N-tiles of 16
#define BN_EPS   1e-5f

typedef __attribute__((ext_vector_type(16))) __bf16 v16bf;
typedef __attribute__((ext_vector_type(8)))  float  v8f;
typedef __attribute__((ext_vector_type(4)))  unsigned int v4u;

union Frag16 { v4u q[2]; v16bf v; };

// ---------------------------------------------------------------------------
// GEMM: C[M,Nout] = A[M,K](bf16) * B[Npad,K](bf16)^T (+bias).
// Per wave: one 16-row strip x NT 16-col tiles via V_WMMA_F32_16X16X32_BF16.
// 4 waves / 128-thread block share the same N-tiles -> B slab staged once in
// LDS via GLOBAL_LOAD_ASYNC_TO_LDS_B128 (ASYNCcnt) and re-read as ds_load_b128.
// 16-bit operand layout (ISA 7.12.2): lane l holds row (l&15); bf16 elements
// [kLane .. kLane+7] and [kLane+16 .. kLane+23] are contiguous -> 2x b128.
// f32 C layout: VGPR r -> row r + (l>=16 ? 8:0), col = l&15.
// ---------------------------------------------------------------------------
template<int NT, bool OUT_BF16>
__global__ __launch_bounds__(128)
void gemm_wmma_bf16(const __bf16* __restrict__ A,
                    const __bf16* __restrict__ B,
                    void* __restrict__ C,
                    const float* __restrict__ bias,
                    int M, int K, int Nout)
{
  __shared__ __attribute__((aligned(16))) __bf16 Bs[NT * 16 * 256]; // K <= 256

  const int lane  = threadIdx.x & 31;
  const int mTile = blockIdx.x * 4 + (threadIdx.x >> 5);
  const int n0    = blockIdx.y * NT * 16;
  const bool active = (mTile * 16 < M);

  // ---- async-stage the B slab (NT*16 rows x K, contiguous) into LDS ----
  {
    const int chunks = NT * 16 * K / 8;                 // 16-byte chunks
    const __bf16* gB = B + (size_t)n0 * K;
    const unsigned lbase = (unsigned)(uintptr_t)(void*)Bs; // low 32b = LDS offset
    for (int c = threadIdx.x; c < chunks; c += 128) {
      unsigned long long ga = (unsigned long long)(uintptr_t)(gB + (size_t)c * 8);
      unsigned la = lbase + (unsigned)c * 16u;
      asm volatile("global_load_async_to_lds_b128 %0, %1, off"
                   :: "v"(la), "v"(ga) : "memory");
    }
    asm volatile("s_wait_asynccnt 0x0" ::: "memory");
  }
  __syncthreads();

  v8f zero = {};
  v8f acc[NT];
#pragma unroll
  for (int t = 0; t < NT; ++t) acc[t] = zero;

  if (active) {
    const int rA    = mTile * 16 + (lane & 15);
    const size_t baseA = (size_t)rA * K;
    const int kLane = (lane >> 4) << 3;                 // 0 or 8
    const int rBloc = lane & 15;

    for (int kk = 0; kk < K; kk += 32) {
      if (kk + 32 < K)                                  // stream-prefetch A
        __builtin_prefetch((const char*)A + ((baseA + kk + 32) << 1), 0, 1);
      Frag16 fa;
      fa.q[0] = *(const v4u*)(A + baseA + kk + kLane);
      fa.q[1] = *(const v4u*)(A + baseA + kk + kLane + 16);
#pragma unroll
      for (int t = 0; t < NT; ++t) {
        Frag16 fb;
        const __bf16* bs = Bs + (size_t)(t * 16 + rBloc) * K + kk + kLane;
        fb.q[0] = *(const v4u*)bs;
        fb.q[1] = *(const v4u*)(bs + 16);
        acc[t] = __builtin_amdgcn_wmma_f32_16x16x32_bf16(false, fa.v, false, fb.v,
                                                         (short)0, acc[t], false, false);
      }
    }

    const int row0 = mTile * 16 + ((lane >> 4) << 3);
#pragma unroll
    for (int t = 0; t < NT; ++t) {
      const int col = n0 + t * 16 + (lane & 15);
      if (col < Nout) {
        const float bb = bias ? bias[col] : 0.0f;
#pragma unroll
        for (int r = 0; r < 8; ++r) {
          const float val = acc[t][r] + bb;
          const size_t idx = (size_t)(row0 + r) * Nout + col;
          if (OUT_BF16) ((__bf16*)C)[idx] = (__bf16)val;
          else          ((float*)C)[idx]  = val;
        }
      }
    }
  }
}

// ------------------------- degree / normalization --------------------------
__global__ void k_init_deg(int* deg) {
  int i = blockIdx.x * blockDim.x + threadIdx.x;
  if (i < NN) deg[i] = 1;                      // self loop
}
__global__ void k_count_deg(const int* __restrict__ dst, int* deg) {
  int e = blockIdx.x * blockDim.x + threadIdx.x;
  if (e < NE) atomicAdd(&deg[dst[e]], 1);
}
__global__ void k_dinv(const int* __restrict__ deg, float* dinv) {
  int i = blockIdx.x * blockDim.x + threadIdx.x;
  if (i < NN) dinv[i] = rsqrtf((float)deg[i]);
}

// ------------------------- GCN aggregation ---------------------------------
__global__ void k_agg_init(const float* __restrict__ hw, const float* __restrict__ dinv,
                           const float* __restrict__ bias, float* __restrict__ out) {
  int i = blockIdx.x * blockDim.x + threadIdx.x;
  if (i >= NN * DHID) return;
  int n = i >> 7, k = i & 127;
  float di = dinv[n];
  out[i] = hw[i] * di * di + bias[k];          // self loop + bias
}
__global__ void k_agg_scatter(const float* __restrict__ hw, const float* __restrict__ dinv,
                              const int* __restrict__ src, const int* __restrict__ dst,
                              float* __restrict__ out) {
  long i = (long)blockIdx.x * blockDim.x + threadIdx.x;
  if (i >= (long)NE * DHID) return;
  int e = (int)(i >> 7), k = (int)(i & 127);
  int s = src[e], d = dst[e];
  float c = dinv[s] * dinv[d];
  atomicAdd(&out[(size_t)d * DHID + k], hw[(size_t)s * DHID + k] * c);
}

// ------------------------- batch-norm machinery ----------------------------
__global__ void k_zero(float* p, int n) {
  int i = blockIdx.x * blockDim.x + threadIdx.x;
  if (i < n) p[i] = 0.0f;
}
__global__ void k_colstats_f32(const float* __restrict__ X, int M, int rowsPer,
                               float* __restrict__ sum, float* __restrict__ sumsq) {
  const int C = blockDim.x;
  int c = threadIdx.x;
  int r0 = blockIdx.x * rowsPer, r1 = r0 + rowsPer;
  if (r1 > M) r1 = M;
  float s = 0.f, ss = 0.f;
  for (int r = r0; r < r1; ++r) {
    float v = X[(size_t)r * C + c];
    s += v; ss += v * v;
  }
  atomicAdd(&sum[c], s); atomicAdd(&sumsq[c], ss);
}
__global__ void k_colstats_bf16(const __bf16* __restrict__ X, int M, int rowsPer,
                                float* __restrict__ sum, float* __restrict__ sumsq) {
  const int C = blockDim.x;
  int c = threadIdx.x;
  int r0 = blockIdx.x * rowsPer, r1 = r0 + rowsPer;
  if (r1 > M) r1 = M;
  float s = 0.f, ss = 0.f;
  for (int r = r0; r < r1; ++r) {
    float v = (float)X[(size_t)r * C + c];
    s += v; ss += v * v;
  }
  atomicAdd(&sum[c], s); atomicAdd(&sumsq[c], ss);
}
__global__ void k_bn_finalize(const float* __restrict__ sum, const float* __restrict__ sumsq,
                              const float* __restrict__ g, const float* __restrict__ be,
                              float invM, int C, float* scale, float* shift) {
  int c = blockIdx.x * blockDim.x + threadIdx.x;
  if (c >= C) return;
  float m  = sum[c] * invM;
  float v  = sumsq[c] * invM - m * m;
  float sc = g[c] * rsqrtf(v + BN_EPS);
  scale[c] = sc;
  shift[c] = be[c] - m * sc;
}

// ------------------------- fused BN/ReLU applies ---------------------------
__global__ void k_bn_relu_node1(const float* __restrict__ agg, const float* __restrict__ sc,
                                const float* __restrict__ sh, float* __restrict__ x1,
                                __bf16* __restrict__ x1b) {
  int i = blockIdx.x * blockDim.x + threadIdx.x;
  if (i >= NN * DHID) return;
  int k = i & 127;
  float v = agg[i] * sc[k] + sh[k];
  v = v > 0.f ? v : 0.f;
  x1[i] = v; x1b[i] = (__bf16)v;
}
__global__ void k_bn_relu_res(const float* __restrict__ agg, const float* __restrict__ sc,
                              const float* __restrict__ sh, const float* __restrict__ x1,
                              float* __restrict__ h) {
  int i = blockIdx.x * blockDim.x + threadIdx.x;
  if (i >= NN * DHID) return;
  int k = i & 127;
  float v = agg[i] * sc[k] + sh[k] + x1[i];
  h[i] = v > 0.f ? v : 0.f;
}
// 8 bf16 per thread (16B load + 16B store)
__global__ void k_bn_relu_edge(const __bf16* __restrict__ in, const float* __restrict__ sc,
                               const float* __restrict__ sh, __bf16* __restrict__ out) {
  long i = (long)blockIdx.x * blockDim.x + threadIdx.x;
  if (i >= (long)NE * (DEDG / 8)) return;
  const int cbase = (int)((i & 31) << 3);
  union { v4u q; __bf16 b[8]; } vi, vo;
  vi.q = *(const v4u*)(in + i * 8);
#pragma unroll
  for (int q = 0; q < 8; ++q) {
    float v = (float)vi.b[q] * sc[cbase + q] + sh[cbase + q];
    vo.b[q] = (__bf16)(v > 0.f ? v : 0.f);
  }
  *(v4u*)(out + i * 8) = vo.q;
}

// ------------------------- gather / converts -------------------------------
// 8 features per thread: 32 threads per edge cover concat([h[src],h[dst]])
__global__ void k_gather_concat(const float* __restrict__ h, const int* __restrict__ src,
                                const int* __restrict__ dst, __bf16* __restrict__ ef) {
  long i = (long)blockIdx.x * blockDim.x + threadIdx.x;
  if (i >= (long)NE * (DEDG / 8)) return;
  int e  = (int)(i >> 5);
  int j8 = (int)(i & 31);
  int n  = (j8 < 16) ? src[e] : dst[e];
  const float* hp = h + (size_t)n * DHID + (size_t)(j8 & 15) * 8;
  union { v4u q; __bf16 b[8]; } vo;
#pragma unroll
  for (int q = 0; q < 8; ++q) vo.b[q] = (__bf16)hp[q];
  *(v4u*)(ef + i * 8) = vo.q;
}
__global__ void k_f32_to_bf16_pad(const float* __restrict__ w, __bf16* __restrict__ o,
                                  int rows, int cols, int outRows) {
  long i = (long)blockIdx.x * blockDim.x + threadIdx.x;
  if (i >= (long)outRows * cols) return;
  int r = (int)(i / cols);
  o[i] = (r < rows) ? (__bf16)w[i] : (__bf16)0.0f;
}

// ===========================================================================
extern "C" void kernel_launch(void* const* d_in, const int* in_sizes, int n_in,
                              void* d_out, int out_size, void* d_ws, size_t ws_size,
                              hipStream_t stream) {
  (void)in_sizes; (void)n_in; (void)out_size; (void)ws_size;

  const float* x   = (const float*)d_in[0];
  const int*   ei  = (const int*)  d_in[1];
  const int*   src = ei;
  const int*   dst = ei + NE;
  const float* w0 = (const float*)d_in[2],  *b0 = (const float*)d_in[3];
  const float* g1 = (const float*)d_in[4],  *be1= (const float*)d_in[5];
  const float* w2 = (const float*)d_in[6],  *b2 = (const float*)d_in[7];
  const float* g3 = (const float*)d_in[8],  *be3= (const float*)d_in[9];
  const float* w4 = (const float*)d_in[10], *b4 = (const float*)d_in[11];
  const float* g5 = (const float*)d_in[12], *be5= (const float*)d_in[13];
  const float* w6 = (const float*)d_in[14], *b6 = (const float*)d_in[15];
  const float* g7 = (const float*)d_in[16], *be7= (const float*)d_in[17];
  const float* w8 = (const float*)d_in[18], *b8 = (const float*)d_in[19];
  float* out = (float*)d_out;

  // ---- workspace carve-out (~385 MB) ----
  char* ws = (char*)d_ws; size_t off = 0;
  auto take = [&](size_t bytes) -> void* {
    void* p = ws + off; off += (bytes + 255) & ~(size_t)255; return p;
  };
  int*    deg    = (int*)   take((size_t)NN * 4);
  float*  dinv   = (float*) take((size_t)NN * 4);
  float*  bsum   = (float*) take(256 * 4);
  float*  bsq    = (float*) take(256 * 4);
  float*  bscale = (float*) take(256 * 4);
  float*  bshift = (float*) take(256 * 4);
  __bf16* xb     = (__bf16*)take((size_t)NN * DIN  * 2);
  __bf16* w0b    = (__bf16*)take((size_t)DHID * DIN  * 2);
  __bf16* w2b    = (__bf16*)take((size_t)DHID * DHID * 2);
  __bf16* w4b    = (__bf16*)take((size_t)DEDG * DEDG * 2);
  __bf16* w6b    = (__bf16*)take((size_t)DEDG * DEDG * 2);
  __bf16* w8b    = (__bf16*)take((size_t)DOUTP * DEDG * 2);
  float*  hw     = (float*) take((size_t)NN * DHID * 4);   // GEMM out (reused)
  float*  agg    = (float*) take((size_t)NN * DHID * 4);   // aggregation (reused)
  float*  x1     = (float*) take((size_t)NN * DHID * 4);
  __bf16* x1b    = (__bf16*)take((size_t)NN * DHID * 2);
  float*  hres   = (float*) take((size_t)NN * DHID * 4);
  __bf16* efA    = (__bf16*)take((size_t)NE * DEDG * 2);   // ping
  __bf16* efB    = (__bf16*)take((size_t)NE * DEDG * 2);   // pong

  auto cdiv = [](long a, long b) { return (unsigned)((a + b - 1) / b); };
  const dim3 B256(256), B128(128);
  const dim3 GB(128);  // gemm block: 4 waves

  // ---- degrees (indegree + self loop) and D^{-1/2} ----
  k_init_deg <<<cdiv(NN, 256), B256, 0, stream>>>(deg);
  k_count_deg<<<cdiv(NE, 256), B256, 0, stream>>>(dst, deg);
  k_dinv     <<<cdiv(NN, 256), B256, 0, stream>>>(deg, dinv);

  // ---- bf16 conversions (weights stay resident in L2) ----
  k_f32_to_bf16_pad<<<cdiv((long)NN * DIN, 256), B256, 0, stream>>>(x,  xb,  NN,   DIN,  NN);
  k_f32_to_bf16_pad<<<cdiv((long)DHID * DIN, 256), B256, 0, stream>>>(w0, w0b, DHID, DIN,  DHID);
  k_f32_to_bf16_pad<<<cdiv((long)DHID * DHID,256), B256, 0, stream>>>(w2, w2b, DHID, DHID, DHID);
  k_f32_to_bf16_pad<<<cdiv((long)DEDG * DEDG,256), B256, 0, stream>>>(w4, w4b, DEDG, DEDG, DEDG);
  k_f32_to_bf16_pad<<<cdiv((long)DEDG * DEDG,256), B256, 0, stream>>>(w6, w6b, DEDG, DEDG, DEDG);
  k_f32_to_bf16_pad<<<cdiv((long)DOUTP* DEDG,256), B256, 0, stream>>>(w8, w8b, DOUT, DEDG, DOUTP);

  // ================= GCN layer 1 =================
  gemm_wmma_bf16<4, false><<<dim3(cdiv(NN / 16, 4), DHID / 64), GB, 0, stream>>>(
      xb, w0b, hw, nullptr, NN, DIN, DHID);
  k_agg_init   <<<cdiv((long)NN * DHID, 256), B256, 0, stream>>>(hw, dinv, b0, agg);
  k_agg_scatter<<<cdiv((long)NE * DHID, 256), B256, 0, stream>>>(hw, dinv, src, dst, agg);
  k_zero<<<2, B256, 0, stream>>>(bsum, 256);
  k_zero<<<2, B256, 0, stream>>>(bsq, 256);
  k_colstats_f32<<<200, B128, 0, stream>>>(agg, NN, 100, bsum, bsq);
  k_bn_finalize <<<1, B128, 0, stream>>>(bsum, bsq, g1, be1, 1.0f / NN, DHID, bscale, bshift);
  k_bn_relu_node1<<<cdiv((long)NN * DHID, 256), B256, 0, stream>>>(agg, bscale, bshift, x1, x1b);

  // ================= GCN layer 2 (+ residual) =================
  gemm_wmma_bf16<4, false><<<dim3(cdiv(NN / 16, 4), DHID / 64), GB, 0, stream>>>(
      x1b, w2b, hw, nullptr, NN, DHID, DHID);
  k_agg_init   <<<cdiv((long)NN * DHID, 256), B256, 0, stream>>>(hw, dinv, b2, agg);
  k_agg_scatter<<<cdiv((long)NE * DHID, 256), B256, 0, stream>>>(hw, dinv, src, dst, agg);
  k_zero<<<2, B256, 0, stream>>>(bsum, 256);
  k_zero<<<2, B256, 0, stream>>>(bsq, 256);
  k_colstats_f32<<<200, B128, 0, stream>>>(agg, NN, 100, bsum, bsq);
  k_bn_finalize <<<1, B128, 0, stream>>>(bsum, bsq, g3, be3, 1.0f / NN, DHID, bscale, bshift);
  k_bn_relu_res <<<cdiv((long)NN * DHID, 256), B256, 0, stream>>>(agg, bscale, bshift, x1, hres);

  // ================= edge features: gather + concat =================
  k_gather_concat<<<cdiv((long)NE * (DEDG / 8), 256), B256, 0, stream>>>(hres, src, dst, efA);

  // ================= edge MLP layer 1 =================
  gemm_wmma_bf16<4, true><<<dim3(cdiv(NE / 16, 4), DEDG / 64), GB, 0, stream>>>(
      efA, w4b, efB, b4, NE, DEDG, DEDG);
  k_zero<<<2, B256, 0, stream>>>(bsum, 256);
  k_zero<<<2, B256, 0, stream>>>(bsq, 256);
  k_colstats_bf16<<<640, B256, 0, stream>>>(efB, NE, 500, bsum, bsq);
  k_bn_finalize  <<<1, B256, 0, stream>>>(bsum, bsq, g5, be5, 1.0f / NE, DEDG, bscale, bshift);
  k_bn_relu_edge <<<cdiv((long)NE * (DEDG / 8), 256), B256, 0, stream>>>(efB, bscale, bshift, efA);

  // ================= edge MLP layer 2 =================
  gemm_wmma_bf16<4, true><<<dim3(cdiv(NE / 16, 4), DEDG / 64), GB, 0, stream>>>(
      efA, w6b, efB, b6, NE, DEDG, DEDG);
  k_zero<<<2, B256, 0, stream>>>(bsum, 256);
  k_zero<<<2, B256, 0, stream>>>(bsq, 256);
  k_colstats_bf16<<<640, B256, 0, stream>>>(efB, NE, 500, bsum, bsq);
  k_bn_finalize  <<<1, B256, 0, stream>>>(bsum, bsq, g7, be7, 1.0f / NE, DEDG, bscale, bshift);
  k_bn_relu_edge <<<cdiv((long)NE * (DEDG / 8), 256), B256, 0, stream>>>(efB, bscale, bshift, efA);

  // ================= output head: [E,256] x [86,256]^T -> f32 =================
  gemm_wmma_bf16<2, false><<<dim3(cdiv(NE / 16, 4), DOUTP / 32), GB, 0, stream>>>(
      efA, w8b, out, b8, NE, DEDG, DOUT);
}
